// GATNet_38019050504331
// MI455X (gfx1250) — matrix operand order repaired and checked by reading, not compiled, not run on hardware
//
#include <hip/hip_runtime.h>
#include <hip/hip_bf16.h>

typedef __attribute__((ext_vector_type(8)))  float  v8f;
typedef __attribute__((ext_vector_type(16))) __bf16 v16bf;

union Frag { uint4 q[2]; v16bf v; };

#define BN_SCALE 0.9999950000374997f  /* 1/sqrt(1+1e-5) */

__device__ __forceinline__ unsigned short f2bf(float f) {
    unsigned u = __float_as_uint(f);
    unsigned r = (u + 0x7FFFu + ((u >> 16) & 1u)) >> 16;
    return (unsigned short)r;
}
__device__ __forceinline__ unsigned f2ord(float f) {
    unsigned u = __float_as_uint(f);
    return (u & 0x80000000u) ? ~u : (u | 0x80000000u);
}
__device__ __forceinline__ float ord2f(unsigned u) {
    return (u & 0x80000000u) ? __uint_as_float(u & 0x7FFFFFFFu) : __uint_as_float(~u);
}

// ---- pack weight matrix [128 x ncols] row-major f32 -> WMMA-B bf16 tiles ----
// layout: [nt][kt][lane][16] ; lane = (k%32>=16?16:0)+(n%16), p = k%16
__global__ void pack_w(const float* __restrict__ W, unsigned short* __restrict__ out,
                       int ncols, int NT) {
    int t = blockIdx.x * blockDim.x + threadIdx.x;
    int total = 128 * NT * 16;
    if (t >= total) return;
    int n = t % (NT * 16), k = t / (NT * 16);
    float v = (n < ncols) ? W[k * ncols + n] : 0.0f;
    int kt = k >> 5, kk = k & 31, nt = n >> 4, nn = n & 15;
    int lane = ((kk >> 4) << 4) + nn;
    int p = kk & 15;
    out[(((nt * 4 + kt) * 32 + lane) << 4) + p] = f2bf(v);
}

// ---- fused (optional ReLU) + BN + f32->bf16 convert, 4 elems/thread ----
__global__ void conv_bn_bf16(const float* __restrict__ x, const float* __restrict__ g,
                             const float* __restrict__ b, unsigned short* __restrict__ out,
                             int N, int relu) {
    int t = blockIdx.x * blockDim.x + threadIdx.x;     // group of 4
    if (t >= N * 32) return;
    int c0 = (t << 2) & 127;
    float4 v = ((const float4*)x)[t];
    float vv[4] = {v.x, v.y, v.z, v.w};
    ushort4 o;
    unsigned short* op = (unsigned short*)&o;
#pragma unroll
    for (int i = 0; i < 4; ++i) {
        float f = vv[i];
        if (relu) f = fmaxf(f, 0.0f);
        f = f * (g[c0 + i] * BN_SCALE) + b[c0 + i];
        op[i] = f2bf(f);
    }
    ((ushort4*)out)[t] = o;
}

// ---- WMMA GEMM: C[rows x ncols] = A[rows x 128](bf16) @ B(packed, LDS-staged) ----
template <int NT>
__global__ void gemm_bf16(const unsigned short* __restrict__ A,
                          const unsigned short* __restrict__ Bp,
                          const float* __restrict__ bias,
                          float* __restrict__ C,
                          int rows, int ncols, int relu) {
    __shared__ unsigned short Bs[NT * 4 * 32 * 16];    // NT=8 -> 32 KB of 320 KB LDS

    // cooperative block-wide copy of packed B into LDS (all waves participate)
    {
        const uint4* gb = (const uint4*)Bp;
        uint4* sb = (uint4*)Bs;
        const int nq = (NT * 4 * 32 * 16) / 8;         // uint4 count
        for (int i = threadIdx.x; i < nq; i += blockDim.x) sb[i] = gb[i];
    }
    __syncthreads();

    int wave = (blockIdx.x * blockDim.x + threadIdx.x) >> 5;
    int lane = threadIdx.x & 31;
    int rowBase = wave * 16;
    if (rowBase >= rows) return;             // wave-uniform: EXEC stays all-1 for WMMA
    int lm = lane & 15, khalf = lane >> 4;

    v8f acc[NT];
#pragma unroll
    for (int i = 0; i < NT; ++i) acc[i] = (v8f){0,0,0,0,0,0,0,0};

    int r = rowBase + lm;
    if (r >= rows) r = rows - 1;             // clamp: keep all lanes active
    const unsigned short* arow = A + (size_t)r * 128;

#pragma unroll
    for (int kt = 0; kt < 4; ++kt) {
        Frag af;
        const uint4* ap = (const uint4*)(arow + kt * 32 + khalf * 16);
        af.q[0] = ap[0];                     // K = khalf*16 .. +7       -> v0..v3
        af.q[1] = ap[1];                     // K = khalf*16+8 .. +15    -> v4..v7
#pragma unroll
        for (int nt = 0; nt < NT; ++nt) {
            Frag bf_;
            const uint4* bp = (const uint4*)(Bs + (((nt * 4 + kt) * 32 + lane) << 4));
            bf_.q[0] = bp[0];
            bf_.q[1] = bp[1];
            acc[nt] = __builtin_amdgcn_wmma_f32_16x16x32_bf16(
                false, af.v, false, bf_.v, (short)0, acc[nt], false, false);
        }
    }

#pragma unroll
    for (int nt = 0; nt < NT; ++nt) {
        int col = nt * 16 + lm;
        float bv = (bias && col < ncols) ? bias[col] : 0.0f;
#pragma unroll
        for (int i = 0; i < 8; ++i) {
            int row = rowBase + khalf * 8 + i;
            if (row < rows && col < ncols) {
                float v = acc[nt][i] + bv;
                if (relu) v = fmaxf(v, 0.0f);
                C[(size_t)row * ncols + col] = v;
            }
        }
    }
}

// ---- per-node per-head attention dots: a_s[n,h]=<h[n,h,:],att_src[h,:]> ----
__global__ void att_dots(const float* __restrict__ h, const float* __restrict__ att_src,
                         const float* __restrict__ att_dst, float* __restrict__ a_s,
                         float* __restrict__ a_d, int N) {
    int t = blockIdx.x * blockDim.x + threadIdx.x;
    if (t >= N * 4) return;
    int n = t >> 2, hd = t & 3;
    const float* hp = h + (size_t)n * 128 + hd * 32;
    const float* as = att_src + hd * 32;
    const float* ad = att_dst + hd * 32;
    float ss = 0.f, sd = 0.f;
#pragma unroll
    for (int d = 0; d < 32; ++d) { float v = hp[d]; ss += v * as[d]; sd += v * ad[d]; }
    a_s[t] = ss;
    a_d[t] = sd;
}

// ---- init per-layer accumulators: out=bias broadcast, m_ord=0 (== -max), s=0 ----
__global__ void init_layer(float* __restrict__ out, const float* __restrict__ bias,
                           unsigned* __restrict__ m_ord, float* __restrict__ s, int N) {
    int t = blockIdx.x * blockDim.x + threadIdx.x;
    if (t < N * 128) out[t] = bias[t & 127];
    if (t < N * 4) { m_ord[t] = 0u; s[t] = 0.f; }
}

__device__ __forceinline__ void edge_pair(const int* src, const int* dst, int E, int e,
                                          int& si, int& di) {
    if (e < E) { si = src[e]; di = dst[e]; } else { si = di = e - E; }
}

// ---- pass 1: segment max of leaky_relu(a_s[src]+a_d[dst]) over dst ----
__global__ void edge_max(const int* __restrict__ src, const int* __restrict__ dst, int E, int N,
                         const float* __restrict__ a_s, const float* __restrict__ a_d,
                         unsigned* __restrict__ m_ord) {
    int t = blockIdx.x * blockDim.x + threadIdx.x;
    if (t >= (E + N) * 4) return;
    int e = t >> 2, hd = t & 3, si, di;
    edge_pair(src, dst, E, e, si, di);
    float v = a_s[si * 4 + hd] + a_d[di * 4 + hd];
    v = v > 0.f ? v : 0.2f * v;
    atomicMax(&m_ord[di * 4 + hd], f2ord(v));
}

// ---- pass 2: w = exp(e - m[dst]); s[dst] += w; cache w ----
__global__ void edge_sum(const int* __restrict__ src, const int* __restrict__ dst, int E, int N,
                         const float* __restrict__ a_s, const float* __restrict__ a_d,
                         const unsigned* __restrict__ m_ord, float* __restrict__ s,
                         float* __restrict__ wbuf) {
    int t = blockIdx.x * blockDim.x + threadIdx.x;
    if (t >= (E + N) * 4) return;
    int e = t >> 2, hd = t & 3, si, di;
    edge_pair(src, dst, E, e, si, di);
    float v = a_s[si * 4 + hd] + a_d[di * 4 + hd];
    v = v > 0.f ? v : 0.2f * v;
    float w = __expf(v - ord2f(m_ord[di * 4 + hd]));
    wbuf[t] = w;
    atomicAdd(&s[di * 4 + hd], w);
}

// ---- pass 3: out[dst] += alpha * h[src]; one wave per edge, lane = head dim ----
__global__ void edge_aggr(const int* __restrict__ src, const int* __restrict__ dst, int E, int N,
                          const float* __restrict__ wbuf, const float* __restrict__ s,
                          const float* __restrict__ h, float* __restrict__ out) {
    int g = blockIdx.x * blockDim.x + threadIdx.x;
    int e = g >> 5, lane = g & 31;
    if (e >= E + N) return;
    int si, di;
    edge_pair(src, dst, E, e, si, di);
#pragma unroll
    for (int hd = 0; hd < 4; ++hd) {
        float alpha = wbuf[e * 4 + hd] / (s[di * 4 + hd] + 1e-16f);
        float v = alpha * h[(size_t)si * 128 + hd * 32 + lane];
        atomicAdd(&out[(size_t)di * 128 + hd * 32 + lane], v);
    }
}

extern "C" void kernel_launch(void* const* d_in, const int* in_sizes, int n_in,
                              void* d_out, int out_size, void* d_ws, size_t ws_size,
                              hipStream_t stream) {
    const int N = in_sizes[0] / 128;
    const int E = in_sizes[1] / 2;
    const float* x = (const float*)d_in[0];
    const int* ei = (const int*)d_in[1];
    const int* src = ei;
    const int* dst = ei + E;

    // workspace carve-out (256B aligned)
    char* wp = (char*)d_ws;
    auto take = [&](size_t bytes) -> void* {
        void* p = (void*)wp;
        wp += (bytes + 255) & ~(size_t)255;
        return p;
    };
    unsigned short* Abf  = (unsigned short*)take((size_t)N * 128 * 2);
    float*          buf0 = (float*)take((size_t)N * 128 * 4);
    float*          buf1 = (float*)take((size_t)N * 128 * 4);
    float*          a_s  = (float*)take((size_t)N * 4 * 4);
    float*          a_d  = (float*)take((size_t)N * 4 * 4);
    unsigned*       mord = (unsigned*)take((size_t)N * 4 * 4);
    float*          ssum = (float*)take((size_t)N * 4 * 4);
    float*          wbuf = (float*)take((size_t)(E + N) * 4 * 4);
    unsigned short* pw[6];
    for (int i = 0; i < 6; ++i) pw[i] = (unsigned short*)take(128 * 128 * 2);

    // weights: W_feat, gat[0..2].W, W_fc, W_cls
    const float* Wm[6]  = { (const float*)d_in[4],  (const float*)d_in[8],
                            (const float*)d_in[14], (const float*)d_in[20],
                            (const float*)d_in[26], (const float*)d_in[30] };
    const int wcols[6] = {128, 128, 128, 128, 128, 40};
    const int wNT[6]   = {8, 8, 8, 8, 8, 3};
    for (int i = 0; i < 6; ++i) {
        int total = 128 * wNT[i] * 16;
        pack_w<<<(total + 255) / 256, 256, 0, stream>>>(Wm[i], pw[i], wcols[i], wNT[i]);
    }

    const int cgrid  = (N * 32 + 255) / 256;           // conv: 4 elems/thread
    const int igrid  = (N * 128 + 255) / 256;
    const int ggrid  = (((N + 15) / 16) * 32 + 255) / 256;
    const int egrid  = ((E + N) * 4 + 255) / 256;
    const int agrid  = ((E + N) * 32 + 255) / 256;
    const int ngrid4 = (N * 4 + 255) / 256;

    // stage 1: x -> BN_feat -> bf16 -> GEMM(W_feat)+b+ReLU -> buf0
    conv_bn_bf16<<<cgrid, 256, 0, stream>>>(x, (const float*)d_in[2], (const float*)d_in[3],
                                            Abf, N, 0);
    gemm_bf16<8><<<ggrid, 256, 0, stream>>>(Abf, pw[0], (const float*)d_in[5], buf0, N, 128, 1);

    // 3 GAT layers
    for (int L = 0; L < 3; ++L) {
        int pb = 6 + 6 * L;
        const float* bng  = (const float*)d_in[pb + 0];
        const float* bnb  = (const float*)d_in[pb + 1];
        const float* atts = (const float*)d_in[pb + 3];
        const float* attd = (const float*)d_in[pb + 4];
        const float* bias = (const float*)d_in[pb + 5];
        // x(buf0) -> [ReLU] -> BN -> bf16
        conv_bn_bf16<<<cgrid, 256, 0, stream>>>(buf0, bng, bnb, Abf, N, (L > 0) ? 1 : 0);
        // h = x_bn @ W  (no bias, no relu)
        gemm_bf16<8><<<ggrid, 256, 0, stream>>>(Abf, pw[1 + L], nullptr, buf1, N, 128, 0);
        att_dots<<<ngrid4, 256, 0, stream>>>(buf1, atts, attd, a_s, a_d, N);
        init_layer<<<igrid, 256, 0, stream>>>(buf0, bias, mord, ssum, N);
        edge_max<<<egrid, 256, 0, stream>>>(src, dst, E, N, a_s, a_d, mord);
        edge_sum<<<egrid, 256, 0, stream>>>(src, dst, E, N, a_s, a_d, mord, ssum, wbuf);
        edge_aggr<<<agrid, 256, 0, stream>>>(src, dst, E, N, wbuf, ssum, buf1, buf0);
        // buf0 now holds bias + sum (pre-ReLU); ReLU folded into next conv
    }

    // head: ReLU -> BN_fc -> GEMM(W_fc)+b+ReLU -> BN_h -> GEMM(W_cls)+b -> d_out
    conv_bn_bf16<<<cgrid, 256, 0, stream>>>(buf0, (const float*)d_in[24], (const float*)d_in[25],
                                            Abf, N, 1);
    gemm_bf16<8><<<ggrid, 256, 0, stream>>>(Abf, pw[4], (const float*)d_in[27], buf1, N, 128, 1);
    conv_bn_bf16<<<cgrid, 256, 0, stream>>>(buf1, (const float*)d_in[28], (const float*)d_in[29],
                                            Abf, N, 0);
    gemm_bf16<3><<<ggrid, 256, 0, stream>>>(Abf, pw[5], (const float*)d_in[31], (float*)d_out,
                                            N, 40, 0);
}
